// CausalSelfAttention_83837761618051
// MI455X (gfx1250) — matrix operand order repaired and checked by reading, not compiled, not run on hardware
//
#include <hip/hip_runtime.h>

// Problem constants (match reference)
constexpr int Bc = 4, Tc = 2048, Cc = 1024, Hc = 16, Dh = 64;
constexpr int C3 = 3 * Cc;

typedef __attribute__((ext_vector_type(16))) _Float16 v16h;
typedef __attribute__((ext_vector_type(8)))  float    v8f;

#define WMMA_F32_F16(A, B, Cacc) \
  __builtin_amdgcn_wmma_f32_16x16x32_f16(false, (A), false, (B), (short)0, (Cacc), false, false)

// ---------------------------------------------------------------------------
// Tiled GEMM with bias: Cout[M,N] = A[M,K] @ Bw[K,N] + bias[N]
// 256 threads = 8 waves; block tile 128(M) x 128(N); K-step 32.
// Wave grid 4(M) x 2(N): each wave computes 2 row-strips x 4 col-tiles
// (8 WMMA per k-step). Global->register prefetch pipelines the next k-step
// under the WMMA block. TOut = float (proj) or _Float16 (qkv intermediate).
//   A frag (16x32 f16):  K = laneHalf*16 + e
//   B frag (32x16 f16):  K = laneHalf*16 + e, N = lane&15
// ---------------------------------------------------------------------------
template <typename TOut>
__global__ __launch_bounds__(256) void gemm_bias_wmma(
    const float* __restrict__ A, const float* __restrict__ Bw,
    const float* __restrict__ bias, TOut* __restrict__ Cout,
    int M, int N, int K)
{
  __shared__ _Float16 lA[8 * 32 * 16];   // [mStrip][lane][e]  (128x32 tile)
  __shared__ _Float16 lB[8 * 32 * 16];   // [nStrip][lane][e]  (32x128 tile)

  const int tid      = threadIdx.x;
  const int lane     = tid & 31;
  const int w        = tid >> 5;       // 0..7
  const int wm       = w & 3;          // owns M strips 2*wm, 2*wm+1
  const int wn       = w >> 2;         // owns N cols [wn*64, wn*64+64)
  const int laneHalf = lane >> 4;
  const int lan15    = lane & 15;
  const int m0       = blockIdx.y * 128;
  const int n0       = blockIdx.x * 128;

  // Per-thread staging sources (thread == one [strip][lane] fragment unit)
  const int    sStrip = tid >> 5;                      // 0..7
  const float* aSrc   = &A[(size_t)(m0 + sStrip * 16 + lan15) * K + (laneHalf << 4)];
  const int    bCol   = n0 + sStrip * 16 + lan15;
  const int    bRow0  = laneHalf << 4;

  v8f acc[2][4] = {};
  float pa[16], pb[16];

  // ---- prologue: prefetch k0 = 0 tile into registers ----
  #pragma unroll
  for (int e = 0; e < 16; ++e) {
    pa[e] = aSrc[e];
    pb[e] = Bw[(size_t)(bRow0 + e) * N + bCol];
  }

  for (int k0 = 0; k0 < K; k0 += 32) {
    // ---- commit prefetched tile to LDS in fragment layout ----
    #pragma unroll
    for (int e = 0; e < 16; ++e) lA[tid * 16 + e] = (_Float16)pa[e];
    #pragma unroll
    for (int e = 0; e < 16; ++e) lB[tid * 16 + e] = (_Float16)pb[e];
    __syncthreads();

    // ---- prefetch next k-step (issues VMEM before the WMMA block) ----
    if (k0 + 32 < K) {
      const int kn = k0 + 32;
      #pragma unroll
      for (int e = 0; e < 16; ++e) {
        pa[e] = aSrc[kn + e];
        pb[e] = Bw[(size_t)(kn + bRow0 + e) * N + bCol];
      }
      if (k0 + 64 < K) {   // L2 prefetch two steps ahead (global_prefetch_b8)
        __builtin_prefetch(&aSrc[k0 + 64], 0, 1);
        __builtin_prefetch(&Bw[(size_t)(k0 + 64 + bRow0) * N + bCol], 0, 1);
      }
    }

    // ---- compute: 8 WMMA per wave per k-step ----
    v16h af0 = *(const v16h*)&lA[((2 * wm + 0) * 32 + lane) * 16];
    v16h af1 = *(const v16h*)&lA[((2 * wm + 1) * 32 + lane) * 16];
    #pragma unroll
    for (int t = 0; t < 4; ++t) {
      v16h bf = *(const v16h*)&lB[((wn * 4 + t) * 32 + lane) * 16];
      acc[0][t] = WMMA_F32_F16(af0, bf, acc[0][t]);
      acc[1][t] = WMMA_F32_F16(af1, bf, acc[1][t]);
    }
    __syncthreads();
  }

  // ---- epilogue: C layout row M = laneHalf*8 + r, col N = lane&15 ----
  #pragma unroll
  for (int s = 0; s < 2; ++s) {
    #pragma unroll
    for (int t = 0; t < 4; ++t) {
      int col  = n0 + wn * 64 + t * 16 + lan15;
      float bv = bias[col];
      #pragma unroll
      for (int r = 0; r < 8; ++r) {
        int row = m0 + (2 * wm + s) * 16 + (laneHalf << 3) + r;
        Cout[(size_t)row * N + col] = (TOut)(acc[s][t][r] + bv);
      }
    }
  }
}

// ---------------------------------------------------------------------------
// Flash attention: one workgroup (128 thr = 4 waves) per (b, h, 64-row qblock).
// qkv is f16 [B,T,3C]. K tiles are brought in with async global->LDS B128
// copies (fragment layout is byte-contiguous per 16-byte chunk); V tiles are
// synchronous f16 copies. Online softmax in f32; S and PV via WMMA.
// ---------------------------------------------------------------------------
__global__ __launch_bounds__(128) void flash_attn_wmma(
    const _Float16* __restrict__ qkv, float* __restrict__ y)
{
  __shared__ _Float16 lK[2 * 4 * 32 * 16];   // K-block in B-frag layout (S = Q K^T)
  __shared__ _Float16 lV[2 * 4 * 32 * 16];   // V-block in B-frag layout (O = P V)
  __shared__ _Float16 lP[4][2 * 32 * 16];    // per-wave P in A-frag layout

  const int tid      = threadIdx.x;
  const int lane     = tid & 31;
  const int w        = tid >> 5;
  const int laneHalf = lane >> 4;
  const int lan15    = lane & 15;

  const int qb = blockIdx.x;   // q block of 64 rows
  const int h  = blockIdx.y;
  const int b  = blockIdx.z;

  const size_t rowBase = (size_t)(b * Tc) * C3;
  const int    h64     = h * Dh;
  const int    qrow0   = qb * 64 + w * 16;

  // Q fragments (16x64 as 2x 16x32), softmax scale 1/8 folded in (exact in f16)
  v16h qf[2];
  #pragma unroll
  for (int kk = 0; kk < 2; ++kk) {
    #pragma unroll
    for (int e = 0; e < 16; ++e) {
      int d = kk * 32 + (laneHalf << 4) + e;
      qf[kk][e] = qkv[rowBase + (size_t)(qrow0 + lan15) * C3 + h64 + d]
                  * (_Float16)0.125f;
    }
  }

  v8f o[4] = {};
  float mrow[8], lrow[8];
  #pragma unroll
  for (int r = 0; r < 8; ++r) { mrow[r] = -3.0e38f; lrow[r] = 0.0f; }

  for (int j = 0; j <= qb; ++j) {      // causal: key blocks up to q block
    // ---- K tile: async global->LDS, 16B chunks in fragment layout ----
    // 8192 bytes = 512 chunks; 128 lanes x 4 rounds.
    #pragma unroll
    for (int rnd = 0; rnd < 4; ++rnd) {
      int c    = rnd * 128 + tid;
      int unit = c >> 1, part = c & 1;           // unit = [kk][p][lane]
      int la = unit & 31, p = (unit >> 5) & 3, kk = unit >> 7;
      int key = p * 16 + (la & 31 & 15);
      int d0  = kk * 32 + ((la >> 4) << 4) + part * 8;
      const _Float16* gsrc =
          &qkv[rowBase + (size_t)(j * 64 + key) * C3 + Cc + h64 + d0];
      unsigned ldsAddr = (unsigned)(size_t)&lK[unit * 16 + part * 8];
      unsigned long long ga = (unsigned long long)(size_t)gsrc;
      asm volatile("global_load_async_to_lds_b128 %0, %1, off"
                   :: "v"(ldsAddr), "v"(ga) : "memory");
    }
    // ---- V tile: synchronous f16 copies in B-frag layout ----
    #pragma unroll
    for (int uu = 0; uu < 2; ++uu) {
      int unit = tid * 2 + uu;                  // 0..255 = [kk][p][lane]
      int la = unit & 31, p = (unit >> 5) & 3, kk = unit >> 7;
      int d = p * 16 + (la & 15);
      const _Float16* src =
          &qkv[rowBase + (size_t)(j * 64 + kk * 32 + ((la >> 4) << 4)) * C3
               + 2 * Cc + h64 + d];
      #pragma unroll
      for (int e = 0; e < 16; ++e)
        lV[unit * 16 + e] = src[(size_t)e * C3];
    }
    asm volatile("s_wait_asynccnt 0" ::: "memory");   // K tile landed in LDS
    __syncthreads();

    // ---- S = Q K^T (scaled): 8 WMMA ----
    v8f s[4] = {};
    #pragma unroll
    for (int kk = 0; kk < 2; ++kk) {
      #pragma unroll
      for (int t = 0; t < 4; ++t) {
        v16h bf = *(const v16h*)&lK[((kk * 4 + t) * 32 + lane) * 16];
        s[t] = WMMA_F32_F16(qf[kk], bf, s[t]);
      }
    }

    // ---- causal mask on the diagonal block (uniform branch) ----
    if (j == qb) {
      #pragma unroll
      for (int t = 0; t < 4; ++t) {
        int key = j * 64 + t * 16 + lan15;
        #pragma unroll
        for (int r = 0; r < 8; ++r) {
          int qr = qrow0 + (laneHalf << 3) + r;
          if (key > qr) s[t][r] = -3.0e38f;
        }
      }
    }

    // ---- online softmax (rows live across each 16-lane half) ----
    #pragma unroll
    for (int r = 0; r < 8; ++r) {
      float mx = fmaxf(fmaxf(s[0][r], s[1][r]), fmaxf(s[2][r], s[3][r]));
      #pragma unroll
      for (int off = 1; off <= 8; off <<= 1)
        mx = fmaxf(mx, __shfl_xor(mx, off, 32));
      float mnew  = fmaxf(mrow[r], mx);
      float alpha = __expf(mrow[r] - mnew);
      mrow[r] = mnew;
      float rs = 0.0f;
      #pragma unroll
      for (int t = 0; t < 4; ++t) {
        float p = __expf(s[t][r] - mnew);
        s[t][r] = p;
        rs += p;
      }
      #pragma unroll
      for (int off = 1; off <= 8; off <<= 1)
        rs += __shfl_xor(rs, off, 32);
      lrow[r] = lrow[r] * alpha + rs;
      #pragma unroll
      for (int t = 0; t < 4; ++t) o[t][r] *= alpha;
    }

    // ---- scatter P (f16) into per-wave LDS in A-fragment layout ----
    #pragma unroll
    for (int t = 0; t < 4; ++t) {
      #pragma unroll
      for (int r = 0; r < 8; ++r) {
        int Mr  = (laneHalf << 3) + r;
        int key = t * 16 + lan15;
        int kk  = key >> 5, k32 = key & 31;
        int e   = k32 & 15;
        int l2  = ((k32 >> 4) & 1) * 16 + Mr;
        lP[w][(kk * 32 + l2) * 16 + e] = (_Float16)s[t][r];
      }
    }
    asm volatile("s_wait_dscnt 0" ::: "memory");   // P store -> fragment reload

    v16h pf0 = *(const v16h*)&lP[w][(0 * 32 + lane) * 16];
    v16h pf1 = *(const v16h*)&lP[w][(1 * 32 + lane) * 16];

    // ---- O += P V: 8 WMMA ----
    #pragma unroll
    for (int t = 0; t < 4; ++t) {
      v16h v0 = *(const v16h*)&lV[((0 * 4 + t) * 32 + lane) * 16];
      o[t] = WMMA_F32_F16(pf0, v0, o[t]);
      v16h v1 = *(const v16h*)&lV[((1 * 4 + t) * 32 + lane) * 16];
      o[t] = WMMA_F32_F16(pf1, v1, o[t]);
    }
    __syncthreads();
  }

  // ---- epilogue: y[b, qrow, h*64 + d] = O / l ----
  #pragma unroll
  for (int t = 0; t < 4; ++t) {
    #pragma unroll
    for (int r = 0; r < 8; ++r) {
      int qr = qrow0 + (laneHalf << 3) + r;
      int d  = t * 16 + lan15;
      y[((size_t)(b * Tc) + qr) * Cc + h64 + d] = o[t][r] / lrow[r];
    }
  }
}

// ---------------------------------------------------------------------------
// Workspace layout: qkv = B*T*3C f16 (50.3 MB) | y = B*T*C f32 (33.6 MB)
// ---------------------------------------------------------------------------
extern "C" void kernel_launch(void* const* d_in, const int* in_sizes, int n_in,
                              void* d_out, int out_size, void* d_ws, size_t ws_size,
                              hipStream_t stream) {
  const float* x     = (const float*)d_in[0];
  const float* Wqkv  = (const float*)d_in[1];
  const float* bqkv  = (const float*)d_in[2];
  const float* Wproj = (const float*)d_in[3];
  const float* bproj = (const float*)d_in[4];
  float* out = (float*)d_out;

  char*      ws   = (char*)d_ws;
  _Float16*  qkv  = (_Float16*)ws;
  float*     ybuf = (float*)(ws + (size_t)Bc * Tc * C3 * sizeof(_Float16));

  // 1) qkv(f16) = x @ Wqkv + bqkv   (M=8192, N=3072, K=1024)
  dim3 g1(C3 / 128, (Bc * Tc) / 128);
  gemm_bias_wmma<_Float16><<<g1, dim3(256), 0, stream>>>(x, Wqkv, bqkv, qkv,
                                                         Bc * Tc, C3, Cc);

  // 2) flash attention -> ybuf [B,T,C] f32
  dim3 g2(Tc / 64, Hc, Bc);
  flash_attn_wmma<<<g2, dim3(128), 0, stream>>>(qkv, ybuf);

  // 3) out = ybuf @ Wproj + bproj   (M=8192, N=1024, K=1024)
  dim3 g3(Cc / 128, (Bc * Tc) / 128);
  gemm_bias_wmma<float><<<g3, dim3(256), 0, stream>>>(ybuf, Wproj, bproj, out,
                                                      Bc * Tc, Cc, Cc);
}
